// CustomGraphConvolutionLayer_78254304133529
// MI455X (gfx1250) — compile-verified, matrix operand order checked by reading, not run on hardware
//
#include <hip/hip_runtime.h>
#include <hip/hip_bf16.h>

// ---------------------------------------------------------------------------
// RGCN forward for MI455X (gfx1250, wave32, WMMA + async global->LDS staging).
//   N=40000 nodes, E=640000 edges, IN=OUT=128, R=8 relations, B=4 bases.
//   hr = x @ W[r] via v_wmma_f32_16x16x32_bf16; A/B tiles staged into LDS with
//   global_load_async_to_lds_b128 (ASYNCcnt) once per block, 8x WMMA reuse.
//   root path folded in as relation index 8, written straight to d_out.
// ---------------------------------------------------------------------------

typedef __attribute__((ext_vector_type(16))) __bf16 v16bf;
typedef __attribute__((ext_vector_type(8)))  __bf16 bf16x8;
typedef __attribute__((ext_vector_type(8)))  float  v8f;
typedef __attribute__((address_space(3)))    unsigned short lds_ushort;

#define IN_F   128
#define OUT_F  128
#define NREL   8
#define NRELW  9   // 8 relations + root
#define NBAS   4

__device__ __forceinline__ unsigned short f2bf(float f) {
    unsigned int u = __float_as_uint(f);
    // round-to-nearest-even
    unsigned int r = (u + 0x7FFFu + ((u >> 16) & 1u)) >> 16;
    return (unsigned short)r;
}

// async copy of 16 bytes: global (per-lane addr) -> LDS (per-lane byte offset)
__device__ __forceinline__ void async_cp16(unsigned lds_byte_off, const void* gaddr) {
    asm volatile("global_load_async_to_lds_b128 %0, %1, off"
                 :: "v"(lds_byte_off), "v"(gaddr) : "memory");
}
__device__ __forceinline__ void wait_async0() {
    asm volatile("s_wait_asynccnt 0" ::: "memory");
}

// wt[r][o][k] (bf16, k contiguous) : W[r] = sum_b comp[r,b]*basis[b], wt[8]=root^T
__global__ void build_wt_kernel(const float* __restrict__ basis,
                                const float* __restrict__ comp,
                                const float* __restrict__ root,
                                unsigned short* __restrict__ wt) {
    int gid = blockIdx.x * blockDim.x + threadIdx.x;
    if (gid >= NRELW * OUT_F * IN_F) return;
    int k = gid & (IN_F - 1);
    int o = (gid >> 7) & (OUT_F - 1);
    int r = gid >> 14;
    float v;
    if (r < NREL) {
        v = 0.f;
#pragma unroll
        for (int b = 0; b < NBAS; ++b)
            v += comp[r * NBAS + b] * basis[((size_t)b * IN_F + k) * OUT_F + o];
    } else {
        v = root[(size_t)k * OUT_F + o];
    }
    wt[gid] = f2bf(v);
}

__global__ void cvt_x_kernel(const float* __restrict__ x,
                             unsigned short* __restrict__ xb, int n) {
    int gid = blockIdx.x * blockDim.x + threadIdx.x;
    if (gid < n) xb[gid] = f2bf(x[gid]);
}

__global__ void zero_counts_kernel(int* __restrict__ counts, int n) {
    int gid = blockIdx.x * blockDim.x + threadIdx.x;
    if (gid < n) counts[gid] = 0;
}

__global__ void count_edges_kernel(const int* __restrict__ edge_index,
                                   const int* __restrict__ edge_type,
                                   int* __restrict__ counts, int E) {
    int e = blockIdx.x * blockDim.x + threadIdx.x;
    if (e >= E) return;
    int d = edge_index[E + e];
    int r = edge_type[e];
    atomicAdd(&counts[d * NREL + r], 1);
}

// One block (256 thr = 8 waves) -> one (mt, r): 16 rows of x against all of W[r].
// A (16x128) and B (128x128) staged in LDS via async b128 copies; each wave
// computes one 16x16 output tile (ot = wave-in-block) with 4 WMMA K-steps.
// blockIdx = mt * 9 + r;  r==8 is the root path writing out = x@root + bias.
__global__ void hr_wmma_kernel(const unsigned short* __restrict__ xb,   // [N][128] bf16
                               const unsigned short* __restrict__ wt,   // [9][128][128] bf16
                               const float* __restrict__ bias,          // [128]
                               float* __restrict__ hr,                  // [N][8][128]
                               float* __restrict__ outp)                // [N][128]
{
    __shared__ unsigned short Bs[OUT_F * IN_F];  // 32 KB, [o][k]
    __shared__ unsigned short As[16 * IN_F];     //  4 KB, [m][k]

    const int mt = blockIdx.x / NRELW;
    const int r  = blockIdx.x % NRELW;
    const int tid = threadIdx.x;

    const unsigned short* bsrc = wt + (size_t)r * (OUT_F * IN_F);
    const unsigned short* asrc = xb + (size_t)mt * 16 * IN_F;

    // LDS byte offsets of the shared arrays (wave-relative, addrspace(3))
    const unsigned bs_off = (unsigned)(size_t)(lds_ushort*)Bs;
    const unsigned as_off = (unsigned)(size_t)(lds_ushort*)As;

    // B: 2048 x 16B chunks over 256 threads -> 8 async ops each
#pragma unroll
    for (int it = 0; it < 8; ++it) {
        int c = tid + it * 256;
        async_cp16(bs_off + c * 16, bsrc + c * 8);
    }
    // A: 256 x 16B chunks -> 1 async op each
    async_cp16(as_off + tid * 16, asrc + tid * 8);

    wait_async0();
    __syncthreads();

    const int wib  = tid >> 5;        // wave-in-block == ot tile
    const int lane = tid & 31;
    const int half = lane >> 4;       // ISA 16-bit A/B layout: lanes 16-31 -> K+8/K+24
    const int l15  = lane & 15;

    const unsigned short* arow = As + l15 * IN_F;
    const unsigned short* brow = Bs + (wib * 16 + l15) * IN_F;

    v8f c = {};
#pragma unroll
    for (int kc = 0; kc < 4; ++kc) {
        int k0 = kc * 32 + half * 8;
        bf16x8 a1 = *(const bf16x8*)(arow + k0);        // K = k0 .. k0+7
        bf16x8 a2 = *(const bf16x8*)(arow + k0 + 16);   // K = k0+16 .. k0+23
        bf16x8 b1 = *(const bf16x8*)(brow + k0);
        bf16x8 b2 = *(const bf16x8*)(brow + k0 + 16);
        v16bf A, B;
#pragma unroll
        for (int i = 0; i < 8; ++i) {
            A[i] = a1[i]; A[8 + i] = a2[i];
            B[i] = b1[i]; B[8 + i] = b2[i];
        }
        c = __builtin_amdgcn_wmma_f32_16x16x32_bf16(
                /*neg_a=*/false, A, /*neg_b=*/false, B,
                /*c_mod=*/(short)0, c, /*reuse_a=*/false, /*reuse_b=*/false);
    }

    int col = wib * 16 + l15;
    if (r < NREL) {
#pragma unroll
        for (int v = 0; v < 8; ++v) {
            int m = mt * 16 + v + half * 8;              // C/D layout: VGPR v -> row
            hr[(((size_t)m * NREL) + r) * OUT_F + col] = c[v];
        }
    } else {
        float bv = bias[col];
#pragma unroll
        for (int v = 0; v < 8; ++v) {
            int m = mt * 16 + v + half * 8;
            outp[(size_t)m * OUT_F + col] = c[v] + bv;   // init out = x@root + bias
        }
    }
}

// One thread per (edge, feature): out[dst, o] += hr[src, r, o] / counts[dst, r]
__global__ void scatter_msg_kernel(const int* __restrict__ edge_index,
                                   const int* __restrict__ edge_type,
                                   const float* __restrict__ hr,
                                   const int* __restrict__ counts,
                                   float* __restrict__ outp, int E) {
    int e = blockIdx.x * 2 + (threadIdx.x >> 7);
    int o = threadIdx.x & (OUT_F - 1);
    if (e >= E) return;
    int s = edge_index[e];
    int d = edge_index[E + e];
    int r = edge_type[e];
    float inv = 1.0f / (float)counts[d * NREL + r];      // referenced counts >= 1
    float m = hr[(((size_t)s * NREL) + r) * OUT_F + o] * inv;
    atomicAdd(outp + (size_t)d * OUT_F + o, m);
}

__global__ void copy_attr_kernel(const float4* __restrict__ src,
                                 float4* __restrict__ dst, int n4) {
    int gid = blockIdx.x * blockDim.x + threadIdx.x;
    if (gid < n4) dst[gid] = src[gid];
}

extern "C" void kernel_launch(void* const* d_in, const int* in_sizes, int n_in,
                              void* d_out, int out_size, void* d_ws, size_t ws_size,
                              hipStream_t stream) {
    const float* x          = (const float*)d_in[0];
    const int*   edge_index = (const int*)  d_in[1];
    const int*   edge_type  = (const int*)  d_in[2];
    const float* edge_attr  = (const float*)d_in[3];
    const float* basis      = (const float*)d_in[4];
    const float* comp       = (const float*)d_in[5];
    const float* root       = (const float*)d_in[6];
    const float* bias       = (const float*)d_in[7];

    const int N = in_sizes[0] / IN_F;      // 40000 (multiple of 16)
    const int E = in_sizes[2];             // 640000

    // workspace carve-out (256 B aligned)
    char* ws = (char*)d_ws;
    size_t off = 0;
    auto carve = [&](size_t bytes) {
        void* p = ws + off;
        off = (off + bytes + 255) & ~(size_t)255;
        return p;
    };
    unsigned short* wt     = (unsigned short*)carve((size_t)NRELW * OUT_F * IN_F * 2);
    unsigned short* xb     = (unsigned short*)carve((size_t)N * IN_F * 2);
    int*            counts = (int*)           carve((size_t)N * NREL * 4);
    float*          hr     = (float*)         carve((size_t)N * NREL * OUT_F * 4);

    float* out_nodes = (float*)d_out;                     // [N,128]
    float* out_attr  = (float*)d_out + (size_t)N * OUT_F; // [E,32]

    const int TB = 256;

    // 1) relation weights (bf16, transposed [r][o][k])
    {
        int n = NRELW * OUT_F * IN_F;
        build_wt_kernel<<<(n + TB - 1) / TB, TB, 0, stream>>>(basis, comp, root, wt);
    }
    // 2) x -> bf16
    {
        int n = N * IN_F;
        cvt_x_kernel<<<(n + TB - 1) / TB, TB, 0, stream>>>(x, xb, n);
    }
    // 3) per-(dst, rel) edge counts
    {
        int n = N * NREL;
        zero_counts_kernel<<<(n + TB - 1) / TB, TB, 0, stream>>>(counts, n);
        count_edges_kernel<<<(E + TB - 1) / TB, TB, 0, stream>>>(edge_index, edge_type, counts, E);
    }
    // 4) WMMA GEMM: hr[:, 0..7, :] and out = x@root + bias (one block per (mt, r))
    {
        int blocks = (N / 16) * NRELW;     // exact, no tail
        hr_wmma_kernel<<<blocks, TB, 0, stream>>>(xb, wt, bias, hr, out_nodes);
    }
    // 5) gather + mean-normalize + scatter-add
    {
        int blocks = (E + 1) / 2;          // 2 edges per 256-thread block
        scatter_msg_kernel<<<blocks, TB, 0, stream>>>(edge_index, edge_type, hr,
                                                      counts, out_nodes, E);
    }
    // 6) edge embeddings = edge_attr (identity)
    {
        int n4 = E * 32 / 4;
        copy_attr_kernel<<<(n4 + TB - 1) / TB, TB, 0, stream>>>(
            (const float4*)edge_attr, (float4*)out_attr, n4);
    }
}
